// Physical2Regional2Physical_31860067402085
// MI455X (gfx1250) — compile-verified, matrix operand order checked by reading, not run on hardware
//
#include <hip/hip_runtime.h>
#include <hip/hip_bf16.h>
#include <math.h>

// ---------------------------------------------------------------------------
// Physical2Regional2Physical GNN for MI455X (gfx1250, wave32, WMMA).
//
// All MLPs lower to fused bf16 WMMA GEMMs:
//   H = gelu(X @ W1 + b1)            -> gemm1_gelu / gemm1_edge (gather fused)
//   Y = LN(H @ W2 + b2)*g + bn + R   -> gemm2_ln (LayerNorm + optional
//                                       segment-sum atomicAdd fused in epilogue)
// Edge MLPs (din=384) gather their three 128-wide input segments
// (edge latent, src node, dst node) directly inside the GEMM A-loads:
// no concat materialization. Weight panels are staged in LDS
// (ds_load_b128 fragment reads). Tiny input dims (3/4/6) are zero padded to
// K=32 so they also ride the WMMA path.
// ---------------------------------------------------------------------------

typedef __attribute__((ext_vector_type(16))) __bf16 bf16x16;
typedef __attribute__((ext_vector_type(8)))  float  f32x8;

#define B_   2
#define NP_  16384
#define NR_  2048
#define E1_  65536
#define E2_  32768
#define E3_  65536
#define D_   128

#define SWS  136   // LDS weight-panel stride in ushorts (272B, breaks bank period)

// ---------------- small device helpers ----------------
__device__ __forceinline__ unsigned short f2bf(float f) {
  unsigned int u = __builtin_bit_cast(unsigned int, f);
  u += 0x7fffu + ((u >> 16) & 1u);           // round-to-nearest-even
  return (unsigned short)(u >> 16);
}
__device__ __forceinline__ float bf2f(unsigned short h) {
  unsigned int u = ((unsigned int)h) << 16;
  return __builtin_bit_cast(float, u);
}
__device__ __forceinline__ float gelu_tanh(float x) {
  float x3 = x * x * x;
  return 0.5f * x * (1.0f + tanhf(0.7978845608028654f * (x + 0.044715f * x3)));
}

struct __align__(32) BfPack { uint4 lo; uint4 hi; };

// A-fragment load (global). Per ISA 7.12.2 (16-bit A 16x32): lane<16 holds
// K=0..7 & 16..23, lane>=16 holds K=8..15 & 24..31 -> two 16B chunks per lane.
__device__ __forceinline__ bf16x16 load_frag(const unsigned short* __restrict__ row,
                                             int kb, int half) {
  BfPack p;
  p.lo = *reinterpret_cast<const uint4*>(row + kb + half * 8);
  p.hi = *reinterpret_cast<const uint4*>(row + kb + 16 + half * 8);
  return __builtin_bit_cast(bf16x16, p);
}

// B-fragment load from the LDS-staged weight panel (row n = output column n).
__device__ __forceinline__ bf16x16 lds_frag(const unsigned short* s,
                                            int n, int kb, int half) {
  const unsigned short* r = s + n * SWS + kb + half * 8;
  BfPack p;
  p.lo = *reinterpret_cast<const uint4*>(r);
  p.hi = *reinterpret_cast<const uint4*>(r + 16);
  return __builtin_bit_cast(bf16x16, p);
}

// ---------------- weight repack (f32 -> transposed bf16) ----------------
// W1 f32 [K,128] -> W1T bf16 [128, Kpad]   (zero pad K to Kpad)
__global__ void pack_w1t(const float* __restrict__ W, unsigned short* __restrict__ O,
                         int K, int Kpad) {
  int i = blockIdx.x * blockDim.x + threadIdx.x;
  int total = 128 * Kpad;
  if (i >= total) return;
  int n = i / Kpad, k = i - n * Kpad;
  O[i] = (k < K) ? f2bf(W[(size_t)k * 128 + n]) : (unsigned short)0;
}
// W2 f32 [128,128] -> W2T bf16 [128,128]
__global__ void pack_w2t(const float* __restrict__ W, unsigned short* __restrict__ O) {
  int i = blockIdx.x * blockDim.x + threadIdx.x;
  if (i >= 128 * 128) return;
  int n = i >> 7, k = i & 127;
  O[i] = f2bf(W[(size_t)k * 128 + n]);
}

// ---------------- GEMM 1 (generic): H = gelu(X @ W1 + b1) -------------------
// X bf16 [M,K], W1T bf16 [128,K]. Block = 256 threads = 8 waves, 128 rows.
// NOTE: M must be a multiple of 128 (true for every call in this graph).
__global__ __launch_bounds__(256)
void gemm1_gelu(const unsigned short* __restrict__ X,
                const unsigned short* __restrict__ W1T,
                const float* __restrict__ b1,
                unsigned short* __restrict__ H,
                int M, int K) {
  __shared__ unsigned short sW[128 * SWS];   // 34 KB

  const int tid  = threadIdx.x;
  const int lane = tid & 31;
  const int wave = tid >> 5;
  const int m0   = (blockIdx.x * 8 + wave) * 16;
  const int nl   = lane & 15;
  const int half = lane >> 4;

  f32x8 c[8];
#pragma unroll
  for (int t = 0; t < 8; ++t) c[t] = (f32x8)(0.0f);

  const unsigned short* arow = X + (size_t)(m0 + nl) * K;

  for (int kc = 0; kc < K; kc += 128) {
    const int kw = (K - kc < 128) ? (K - kc) : 128;
    const int vecs = kw >> 3;                         // uint4s per row
    for (int cc = tid; cc < 128 * vecs; cc += 256) {
      int n = cc / vecs, j = cc - n * vecs;
      *reinterpret_cast<uint4*>(&sW[n * SWS + j * 8]) =
          *reinterpret_cast<const uint4*>(W1T + (size_t)n * K + kc + j * 8);
    }
    __syncthreads();

    if (kc + 128 < K)
      __builtin_prefetch(arow + kc + 128, 0, 1);

    for (int kb = 0; kb < kw; kb += 32) {
      bf16x16 bfr[8];
#pragma unroll
      for (int t = 0; t < 8; ++t) bfr[t] = lds_frag(sW, t * 16 + nl, kb, half);
      bf16x16 a = load_frag(arow, kc + kb, half);
#pragma unroll
      for (int t = 0; t < 8; ++t)
        c[t] = __builtin_amdgcn_wmma_f32_16x16x32_bf16(false, a, false, bfr[t],
                                                       (short)0, c[t], false, false);
    }
    __syncthreads();
  }

#pragma unroll
  for (int t = 0; t < 8; ++t) {
    const int col = t * 16 + nl;
    const float bias = b1[col];
#pragma unroll
    for (int r = 0; r < 8; ++r) {
      const int row = m0 + r + 8 * half;   // C layout: VGPR r -> M = r + 8*half
      float v = c[t][r] + bias;
      H[(size_t)row * 128 + col] = f2bf(gelu_tanh(v));
    }
  }
}

// ---------------- GEMM 1 (edge, gather-fused): K = 384 ----------------------
// A row = [ e(row) | node_src(src[e]) | node_dst(dst[e]) ], each 128 wide,
// gathered directly in the A-fragment loads (concat never materialized).
__global__ __launch_bounds__(256)
void gemm1_edge(const unsigned short* __restrict__ eb,
                const unsigned short* __restrict__ sb,
                const unsigned short* __restrict__ db,
                const int* __restrict__ src, const int* __restrict__ dst,
                int E, int Ns, int Nd,
                const unsigned short* __restrict__ W1T,   // [128,384]
                const float* __restrict__ b1,
                unsigned short* __restrict__ H, int M) {
  __shared__ unsigned short sW[128 * SWS];

  const int tid  = threadIdx.x;
  const int lane = tid & 31;
  const int wave = tid >> 5;
  const int m0   = (blockIdx.x * 8 + wave) * 16;
  const int nl   = lane & 15;
  const int half = lane >> 4;

  const int row = m0 + nl;
  const int b   = row / E;
  const int e   = row - b * E;
  const unsigned short* segs[3];
  segs[0] = eb + (size_t)row * 128;
  segs[1] = sb + ((size_t)b * Ns + src[e]) * 128;
  segs[2] = db + ((size_t)b * Nd + dst[e]) * 128;

  f32x8 c[8];
#pragma unroll
  for (int t = 0; t < 8; ++t) c[t] = (f32x8)(0.0f);

  for (int ch = 0; ch < 3; ++ch) {
    // stage W1T[:, ch*128 : (ch+1)*128] into LDS
    for (int cc = tid; cc < 128 * 16; cc += 256) {
      int n = cc >> 4, j = cc & 15;
      *reinterpret_cast<uint4*>(&sW[n * SWS + j * 8]) =
          *reinterpret_cast<const uint4*>(W1T + (size_t)n * 384 + ch * 128 + j * 8);
    }
    __syncthreads();

    const unsigned short* arow = segs[ch];
#pragma unroll
    for (int kb = 0; kb < 128; kb += 32) {
      bf16x16 bfr[8];
#pragma unroll
      for (int t = 0; t < 8; ++t) bfr[t] = lds_frag(sW, t * 16 + nl, kb, half);
      bf16x16 a = load_frag(arow, kb, half);
#pragma unroll
      for (int t = 0; t < 8; ++t)
        c[t] = __builtin_amdgcn_wmma_f32_16x16x32_bf16(false, a, false, bfr[t],
                                                       (short)0, c[t], false, false);
    }
    __syncthreads();
  }

#pragma unroll
  for (int t = 0; t < 8; ++t) {
    const int col = t * 16 + nl;
    const float bias = b1[col];
#pragma unroll
    for (int r = 0; r < 8; ++r) {
      const int orow = m0 + r + 8 * half;
      float v = c[t][r] + bias;
      H[(size_t)orow * 128 + col] = f2bf(gelu_tanh(v));
    }
  }
}

// ---------------- GEMM 2: Y = LN(H @ W2 + b2)*g + bn (+res) -----------------
// K fixed at 128. LayerNorm over the 128 output cols done in-register.
// Optionally fuses the edge->node segment sum: agg[b,dst[e],:] += Y[row,:]
// via f32 atomics (saves a full re-read of Y in a separate scatter pass).
__global__ __launch_bounds__(256)
void gemm2_ln(const unsigned short* __restrict__ Hin,
              const unsigned short* __restrict__ W2T,
              const float* __restrict__ b2,
              const float* __restrict__ g,
              const float* __restrict__ bn,
              const float* __restrict__ res, int has_res,
              float* __restrict__ Yf,
              unsigned short* __restrict__ Ybf,
              int M,
              const int* __restrict__ sdst, int sE, int sN,
              float* __restrict__ sagg) {
  __shared__ unsigned short sW[128 * SWS];

  const int tid  = threadIdx.x;
  const int lane = tid & 31;
  const int wave = tid >> 5;
  const int m0   = (blockIdx.x * 8 + wave) * 16;
  const int nl   = lane & 15;
  const int half = lane >> 4;

  // stage W2T (128x128) into LDS
  for (int cc = tid; cc < 128 * 16; cc += 256) {
    int n = cc >> 4, j = cc & 15;
    *reinterpret_cast<uint4*>(&sW[n * SWS + j * 8]) =
        *reinterpret_cast<const uint4*>(W2T + (size_t)n * 128 + j * 8);
  }
  __syncthreads();

  f32x8 c[8];
#pragma unroll
  for (int t = 0; t < 8; ++t) c[t] = (f32x8)(0.0f);

  const unsigned short* arow = Hin + (size_t)(m0 + nl) * 128;
#pragma unroll
  for (int kb = 0; kb < 128; kb += 32) {
    bf16x16 bfr[8];
#pragma unroll
    for (int t = 0; t < 8; ++t) bfr[t] = lds_frag(sW, t * 16 + nl, kb, half);
    bf16x16 a = load_frag(arow, kb, half);
#pragma unroll
    for (int t = 0; t < 8; ++t)
      c[t] = __builtin_amdgcn_wmma_f32_16x16x32_bf16(false, a, false, bfr[t],
                                                     (short)0, c[t], false, false);
  }

  float bias[8], gg[8], bb[8];
#pragma unroll
  for (int t = 0; t < 8; ++t) {
    int col = t * 16 + nl;
    bias[t] = b2[col]; gg[t] = g[col]; bb[t] = bn[col];
  }

#pragma unroll
  for (int r = 0; r < 8; ++r) {
    const int row = m0 + r + 8 * half;
    float v[8];
    float s = 0.0f;
#pragma unroll
    for (int t = 0; t < 8; ++t) { v[t] = c[t][r] + bias[t]; s += v[t]; }
    s += __shfl_xor(s, 1, 32); s += __shfl_xor(s, 2, 32);
    s += __shfl_xor(s, 4, 32); s += __shfl_xor(s, 8, 32);
    float mean = s * (1.0f / 128.0f);
    float q = 0.0f;
#pragma unroll
    for (int t = 0; t < 8; ++t) { float d = v[t] - mean; q += d * d; }
    q += __shfl_xor(q, 1, 32); q += __shfl_xor(q, 2, 32);
    q += __shfl_xor(q, 4, 32); q += __shfl_xor(q, 8, 32);
    float rstd = rsqrtf(q * (1.0f / 128.0f) + 1e-5f);

    float* aggRow = nullptr;
    if (sdst) {
      int b = row / sE;
      int e = row - b * sE;
      aggRow = sagg + ((size_t)b * sN + sdst[e]) * 128;
    }
#pragma unroll
    for (int t = 0; t < 8; ++t) {
      int col = t * 16 + nl;
      float y = (v[t] - mean) * rstd * gg[t] + bb[t];
      if (has_res) y += res[(size_t)row * 128 + col];
      Yf[(size_t)row * 128 + col] = y;
      Ybf[(size_t)row * 128 + col] = f2bf(y);
      if (sdst) atomicAdd(aggRow + col, y);
    }
  }
}

// ---------------- input builders -------------------------------------------
__global__ void build_enc_p(const float* __restrict__ pc, const float* __restrict__ pnd,
                            unsigned short* __restrict__ X) {
  int i = blockIdx.x * blockDim.x + threadIdx.x;
  if (i >= B_ * NP_) return;
  int b = i / NP_, p = i - b * NP_;
  unsigned short* o = X + (size_t)i * 32;
  o[0] = f2bf(pc[p * 3 + 0]); o[1] = f2bf(pc[p * 3 + 1]); o[2] = f2bf(pc[p * 3 + 2]);
  const float* pd = pnd + ((size_t)b * NP_ + p) * 3;
  o[3] = f2bf(pd[0]); o[4] = f2bf(pd[1]); o[5] = f2bf(pd[2]);
#pragma unroll
  for (int k = 6; k < 32; ++k) o[k] = 0;
}

__global__ void build_enc_r(const float* __restrict__ rc, unsigned short* __restrict__ X) {
  int i = blockIdx.x * blockDim.x + threadIdx.x;
  if (i >= B_ * NR_) return;
  int r = i % NR_;
  unsigned short* o = X + (size_t)i * 32;
  o[0] = f2bf(rc[r * 3 + 0]); o[1] = f2bf(rc[r * 3 + 1]); o[2] = f2bf(rc[r * 3 + 2]);
#pragma unroll
  for (int k = 3; k < 32; ++k) o[k] = 0;
}

__global__ void build_edata(const float* __restrict__ ed, int E,
                            unsigned short* __restrict__ X) {
  int i = blockIdx.x * blockDim.x + threadIdx.x;
  if (i >= B_ * E) return;
  int e = i % E;
  unsigned short* o = X + (size_t)i * 32;
  o[0] = f2bf(ed[e * 4 + 0]); o[1] = f2bf(ed[e * 4 + 1]);
  o[2] = f2bf(ed[e * 4 + 2]); o[3] = f2bf(ed[e * 4 + 3]);
#pragma unroll
  for (int k = 4; k < 32; ++k) o[k] = 0;
}

// X[row] = [ node(row) , agg(row) ]  (bf16, 2*128); agg arrives as f32
__global__ __launch_bounds__(128)
void concat2(const unsigned short* __restrict__ nb,
             const float* __restrict__ agg,
             unsigned short* __restrict__ X) {
  int row = blockIdx.x;
  int d = threadIdx.x;
  unsigned short* o = X + (size_t)row * 256;
  o[d]       = nb[(size_t)row * 128 + d];
  o[128 + d] = f2bf(agg[(size_t)row * 128 + d]);
}

__global__ void zero_f32(float* __restrict__ p, size_t n) {
  for (size_t i = blockIdx.x * (size_t)blockDim.x + threadIdx.x; i < n;
       i += (size_t)gridDim.x * blockDim.x) p[i] = 0.0f;
}

// out[row,0:3] = H[row] @ W2[128,3] + b2   (final decoder, no norm)
__global__ void dec_out_final(const unsigned short* __restrict__ H,
                              const float* __restrict__ W2,
                              const float* __restrict__ b2,
                              float* __restrict__ out, int M) {
  int row = blockIdx.x * blockDim.x + threadIdx.x;
  if (row >= M) return;
  float a0 = b2[0], a1 = b2[1], a2 = b2[2];
  const unsigned short* h = H + (size_t)row * 128;
#pragma unroll 4
  for (int k = 0; k < 128; ++k) {
    float hv = bf2f(h[k]);
    a0 += hv * W2[k * 3 + 0];
    a1 += hv * W2[k * 3 + 1];
    a2 += hv * W2[k * 3 + 2];
  }
  out[row * 3 + 0] = a0; out[row * 3 + 1] = a1; out[row * 3 + 2] = a2;
}

// ---------------------------------------------------------------------------
// Host orchestration
// ---------------------------------------------------------------------------
struct Mlp {
  const float *W1, *W2, *b1, *b2, *g, *bn;
  int din, kpad;
  unsigned short *w1t, *w2t;   // packed bf16 (device, in workspace)
};

extern "C" void kernel_launch(void* const* d_in, const int* in_sizes, int n_in,
                              void* d_out, int out_size, void* d_ws, size_t ws_size,
                              hipStream_t stream) {
  (void)in_sizes; (void)n_in; (void)out_size; (void)ws_size;

  // ---- inputs (setup_inputs order; params pytree flattened: sorted dict keys)
  const float* pndata     = (const float*)d_in[0];
  const float* p_coords   = (const float*)d_in[1];
  const float* r_coords   = (const float*)d_in[2];
  const float* p2r_edata  = (const float*)d_in[3];
  const float* r2r_edata  = (const float*)d_in[4];
  const float* r2p_edata  = (const float*)d_in[5];
  // params leaves start at 6; per-MLP leaf order (sorted): W1,W2,b1,b2,bn,g
  // top-level order: dec_e, dec_edge, dec_node, dec_out, dec_r, enc_e,
  //                  enc_edge, enc_node_dst, enc_node_src, enc_p, enc_r,
  //                  proc[0].edge, proc[0].node, ... proc[2].node, proc_e
  const int* p2r_src = (const int*)d_in[114];
  const int* p2r_dst = (const int*)d_in[115];
  const int* r2r_src = (const int*)d_in[116];
  const int* r2r_dst = (const int*)d_in[117];
  const int* r2p_src = (const int*)d_in[118];
  const int* r2p_dst = (const int*)d_in[119];

  const int mlp_base[18] = {6,12,18,24,30,36,42,48,54,60,66,72,78,84,90,96,102,108};
  const int mlp_din[18]  = {4,384,256,128,128,4,384,256,128,6,3,384,256,384,256,384,256,4};
  enum { DEC_E=0, DEC_EDGE=1, DEC_NODE=2, DEC_OUT=3, DEC_R=4, ENC_E=5, ENC_EDGE=6,
         ENC_NODE_DST=7, ENC_NODE_SRC=8, ENC_P=9, ENC_R=10,
         P0E=11, P0N=12, P1E=13, P1N=14, P2E=15, P2N=16, PROC_E=17 };

  Mlp mlp[18];
  size_t ofs = 0;
  auto walloc = [&](size_t bytes) -> char* {
    char* p = (char*)d_ws + ofs;
    ofs += (bytes + 255) & ~(size_t)255;
    return p;
  };

  for (int m = 0; m < 18; ++m) {
    int bse = mlp_base[m];
    mlp[m].W1 = (const float*)d_in[bse + 0];
    mlp[m].W2 = (const float*)d_in[bse + 1];
    mlp[m].b1 = (const float*)d_in[bse + 2];
    mlp[m].b2 = (const float*)d_in[bse + 3];
    mlp[m].bn = (const float*)d_in[bse + 4];
    mlp[m].g  = (const float*)d_in[bse + 5];
    mlp[m].din  = mlp_din[m];
    mlp[m].kpad = (mlp_din[m] < 32) ? 32 : mlp_din[m];
    mlp[m].w1t = (unsigned short*)walloc((size_t)128 * mlp[m].kpad * 2);
    mlp[m].w2t = (m == DEC_OUT) ? nullptr : (unsigned short*)walloc((size_t)128 * 128 * 2);
  }

  // ---- activation workspace
  float*          PN_F  = (float*)walloc((size_t)B_ * NP_ * D_ * 4);
  unsigned short* PN_B  = (unsigned short*)walloc((size_t)B_ * NP_ * D_ * 2);
  float*          RN_F  = (float*)walloc((size_t)B_ * NR_ * D_ * 4);
  unsigned short* RN_B  = (unsigned short*)walloc((size_t)B_ * NR_ * D_ * 2);
  float*          RND_F = (float*)walloc((size_t)B_ * NR_ * D_ * 4);
  unsigned short* RND_B = (unsigned short*)walloc((size_t)B_ * NR_ * D_ * 2);
  float*          E_F   = (float*)walloc((size_t)B_ * E1_ * D_ * 4);   // reused for r2p
  unsigned short* E_B   = (unsigned short*)walloc((size_t)B_ * E1_ * D_ * 2);
  float*          ER_F  = (float*)walloc((size_t)B_ * E2_ * D_ * 4);
  unsigned short* ER_B  = (unsigned short*)walloc((size_t)B_ * E2_ * D_ * 2);
  float*          AGG   = (float*)walloc((size_t)B_ * NP_ * D_ * 4);   // max(NP,NR)
  unsigned short* XBUF  = (unsigned short*)walloc((size_t)B_ * NP_ * 256 * 2);
  unsigned short* HBUF  = (unsigned short*)walloc((size_t)B_ * E1_ * D_ * 2);

  // ---- pack weights (deterministic every call)
  for (int m = 0; m < 18; ++m) {
    int tot1 = 128 * mlp[m].kpad;
    pack_w1t<<<dim3((tot1 + 255) / 256), dim3(256), 0, stream>>>(
        mlp[m].W1, mlp[m].w1t, mlp[m].din, mlp[m].kpad);
    if (mlp[m].w2t)
      pack_w2t<<<dim3((128 * 128 + 255) / 256), dim3(256), 0, stream>>>(
          mlp[m].W2, mlp[m].w2t);
  }

  auto run_mlp = [&](const Mlp& m, const unsigned short* Xb, int M,
                     const float* res, float* Yf, unsigned short* Ybf) {
    gemm1_gelu<<<dim3((M + 127) / 128), dim3(256), 0, stream>>>(
        Xb, m.w1t, m.b1, HBUF, M, m.kpad);
    gemm2_ln<<<dim3((M + 127) / 128), dim3(256), 0, stream>>>(
        HBUF, m.w2t, m.b2, m.g, m.bn, res, res ? 1 : 0, Yf, Ybf, M,
        nullptr, 0, 0, nullptr);
  };
  // Edge layer: gather-fused GEMM1, then GEMM2 with fused segment-sum scatter.
  auto run_edge_mlp = [&](const Mlp& m,
                          const unsigned short* eb, const unsigned short* sb,
                          const unsigned short* db,
                          const int* src, const int* dst,
                          int E, int Ns, int Nd,
                          float* Yf, unsigned short* Ybf, int aggN) {
    int M = B_ * E;
    gemm1_edge<<<dim3(M / 128), dim3(256), 0, stream>>>(
        eb, sb, db, src, dst, E, Ns, Nd, m.w1t, m.b1, HBUF, M);
    zero_f32<<<dim3(2048), dim3(256), 0, stream>>>(AGG, (size_t)B_ * aggN * D_);
    gemm2_ln<<<dim3(M / 128), dim3(256), 0, stream>>>(
        HBUF, m.w2t, m.b2, m.g, m.bn, Yf, 1, Yf, Ybf, M,
        dst, E, aggN, AGG);
  };

  // ===================== encode =====================
  build_enc_p<<<dim3((B_ * NP_ + 255) / 256), dim3(256), 0, stream>>>(p_coords, pndata, XBUF);
  run_mlp(mlp[ENC_P], XBUF, B_ * NP_, nullptr, PN_F, PN_B);

  build_enc_r<<<dim3((B_ * NR_ + 255) / 256), dim3(256), 0, stream>>>(r_coords, XBUF);
  run_mlp(mlp[ENC_R], XBUF, B_ * NR_, nullptr, RN_F, RN_B);

  build_edata<<<dim3((B_ * E1_ + 255) / 256), dim3(256), 0, stream>>>(p2r_edata, E1_, XBUF);
  run_mlp(mlp[ENC_E], XBUF, B_ * E1_, nullptr, E_F, E_B);

  run_edge_mlp(mlp[ENC_EDGE], E_B, PN_B, RN_B, p2r_src, p2r_dst,
               E1_, NP_, NR_, E_F, E_B, NR_);

  concat2<<<dim3(B_ * NR_), dim3(128), 0, stream>>>(RN_B, AGG, XBUF);
  run_mlp(mlp[ENC_NODE_DST], XBUF, B_ * NR_, RN_F, RN_F, RN_B);

  run_mlp(mlp[ENC_NODE_SRC], PN_B, B_ * NP_, PN_F, PN_F, PN_B);

  // ===================== process =====================
  build_edata<<<dim3((B_ * E2_ + 255) / 256), dim3(256), 0, stream>>>(r2r_edata, E2_, XBUF);
  run_mlp(mlp[PROC_E], XBUF, B_ * E2_, nullptr, ER_F, ER_B);

  const int pe[3] = {P0E, P1E, P2E};
  const int pn[3] = {P0N, P1N, P2N};
  for (int s = 0; s < 3; ++s) {
    run_edge_mlp(mlp[pe[s]], ER_B, RN_B, RN_B, r2r_src, r2r_dst,
                 E2_, NR_, NR_, ER_F, ER_B, NR_);

    concat2<<<dim3(B_ * NR_), dim3(128), 0, stream>>>(RN_B, AGG, XBUF);
    run_mlp(mlp[pn[s]], XBUF, B_ * NR_, RN_F, RN_F, RN_B);
  }

  // ===================== decode =====================
  run_mlp(mlp[DEC_R], RN_B, B_ * NR_, nullptr, RND_F, RND_B);

  build_edata<<<dim3((B_ * E3_ + 255) / 256), dim3(256), 0, stream>>>(r2p_edata, E3_, XBUF);
  run_mlp(mlp[DEC_E], XBUF, B_ * E3_, nullptr, E_F, E_B);

  run_edge_mlp(mlp[DEC_EDGE], E_B, RND_B, PN_B, r2p_src, r2p_dst,
               E3_, NR_, NP_, E_F, E_B, NP_);

  concat2<<<dim3(B_ * NP_), dim3(128), 0, stream>>>(PN_B, AGG, XBUF);
  run_mlp(mlp[DEC_NODE], XBUF, B_ * NP_, PN_F, PN_F, PN_B);

  // final projection: gelu layer via WMMA, 128->3 via VALU
  gemm1_gelu<<<dim3((B_ * NP_ + 127) / 128), dim3(256), 0, stream>>>(
      PN_B, mlp[DEC_OUT].w1t, mlp[DEC_OUT].b1, HBUF, B_ * NP_, 128);
  dec_out_final<<<dim3((B_ * NP_ + 255) / 256), dim3(256), 0, stream>>>(
      HBUF, mlp[DEC_OUT].W2, mlp[DEC_OUT].b2, (float*)d_out, B_ * NP_);
}
